// AssemblyNet_59150289600866
// MI455X (gfx1250) — matrix operand order repaired and checked by reading, not compiled, not run on hardware
//
#include <hip/hip_runtime.h>
#include <math.h>

typedef __attribute__((ext_vector_type(16))) __bf16 v16bf;
typedef __attribute__((ext_vector_type(8)))  __bf16 v8bf;
typedef __attribute__((ext_vector_type(8)))  float  v8f;

#define NNODES 50000
#define NEDGES 800000
#define WID    128

// ---------------------------------------------------------------------------
// init: fill node-max scratch with -inf
// ---------------------------------------------------------------------------
__global__ void init_max_kernel(float* __restrict__ p, long long n) {
    long long i = (long long)blockIdx.x * blockDim.x + threadIdx.x;
    long long stride = (long long)gridDim.x * blockDim.x;
    for (; i < n; i += stride) p[i] = -INFINITY;
}

// ---------------------------------------------------------------------------
// Edge kernel: per 16-edge tile
//   diffs = x[dst]-x[src]                     (fp32 in LDS, bf16 A-tile)
//   h     = relu([diffs, ef] @ W_edge + b)    (bf16 WMMA, fp32 accum)
//   e_val = diffs + h ; atomic fmax scatter to nmax[dst]
// Block = 256 threads = 8 waves; wave w owns output columns [16w, 16w+16).
// ---------------------------------------------------------------------------
#define KS_E 136   // padded k-stride (k in [0,130)) for transposed weights / A
__global__ __launch_bounds__(256) void edge_kernel(
    const float* __restrict__ x,  const int* __restrict__ ei,
    const float* __restrict__ ef, const float* __restrict__ We,
    const float* __restrict__ be, float* __restrict__ nmax)
{
    extern __shared__ char smem[];
    __bf16* sWt = (__bf16*)smem;                       // [128][KS_E] W^T bf16
    __bf16* sA  = (__bf16*)(smem + 128 * KS_E * 2);    // [16][KS_E]  diffs bf16
    float*  sD  = (float*)(smem + 128 * KS_E * 2 + 16 * KS_E * 2); // [16][132] fp32
    float*  sEf = sD + 16 * 132;                       // [16][2]
    float*  sB  = sEf + 32;                            // [128] bias
    int*    sDst = (int*)(sB + 128);                   // [16]

    const int tid = threadIdx.x;

    // Stage transposed weights (rows 0..129), coalesced over n
    for (int i = tid; i < 130 * 128; i += 256) {
        int k = i >> 7, n = i & 127;
        sWt[n * KS_E + k] = (__bf16)We[k * 128 + n];
    }
    for (int i = tid; i < 128; i += 256) sB[i] = be[i];
    __syncthreads();

    const int  wave = tid >> 5, lane = tid & 31;
    const int  nloc = lane & 15, half = lane >> 4;
    const int  n    = (wave << 4) + nloc;

    const long long nTiles = (2LL * NEDGES + 15) / 16;
    for (long long t = blockIdx.x; t < nTiles; t += gridDim.x) {
        const long long base = t * 16;

        // ---- gather + diff : 16 edges x 128 features ----
        for (int i = tid; i < 16 * 128; i += 256) {
            int m = i >> 7, c = i & 127;
            long long e = base + m;
            int s = 0, d = 0;
            float diff = 0.f;
            if (e < 2LL * NEDGES) {
                if (e < NEDGES) { s = ei[e];                   d = ei[NEDGES + e]; }
                else            { s = ei[NEDGES + (e-NEDGES)]; d = ei[e - NEDGES]; }
                diff = x[(long long)d * WID + c] - x[(long long)s * WID + c];
            }
            sD[m * 132 + c] = diff;
            sA[m * KS_E + c] = (__bf16)diff;
            if (c == 0) sDst[m] = d;
        }
        if (tid < 32) {
            int m = tid >> 1, j = tid & 1;
            long long e = base + m;
            float v = 0.f;
            if (e < 2LL * NEDGES) {
                long long eo = (e < NEDGES) ? e : (e - NEDGES);
                v = ef[eo * 2 + j];
            }
            sEf[m * 2 + j] = v;
        }
        __syncthreads();

        // ---- accumulator init: bias + edge-feature columns (k=128,129) ----
        float w128 = (float)sWt[n * KS_E + 128];
        float w129 = (float)sWt[n * KS_E + 129];
        v8f c;
        #pragma unroll
        for (int r = 0; r < 8; ++r) {
            int m = r + (half << 3);
            c[r] = sB[n] + sEf[m * 2 + 0] * w128 + sEf[m * 2 + 1] * w129;
        }

        // ---- K=128 via 4 x wmma_f32_16x16x32_bf16 ----
        #pragma unroll
        for (int kk = 0; kk < 4; ++kk) {
            const int kb = kk * 32;
            // A frag: row m = lane&15 ; k = kb + 8*half + {0..7} and +16
            const __bf16* arow = &sA[(lane & 15) * KS_E + kb + (half << 3)];
            v8bf alo = *(const v8bf*)(arow);
            v8bf ahi = *(const v8bf*)(arow + 16);
            v16bf a = __builtin_shufflevector(alo, ahi,
                0,1,2,3,4,5,6,7,8,9,10,11,12,13,14,15);
            // B frag: col n ; k = kb + 8*half + {0..7} and +16
            const __bf16* brow = &sWt[n * KS_E + kb + (half << 3)];
            v8bf blo = *(const v8bf*)(brow);
            v8bf bhi = *(const v8bf*)(brow + 16);
            v16bf b = __builtin_shufflevector(blo, bhi,
                0,1,2,3,4,5,6,7,8,9,10,11,12,13,14,15);
            c = __builtin_amdgcn_wmma_f32_16x16x32_bf16(
                    false, a, false, b, (short)0, c, false, false);
        }

        // ---- epilogue: relu + residual + scatter fmax (fp32) ----
        #pragma unroll
        for (int r = 0; r < 8; ++r) {
            int m = r + (half << 3);
            long long e = base + m;
            if (e < 2LL * NEDGES) {
                float h = c[r];
                h = h > 0.f ? h : 0.f;
                float val = sD[m * 132 + n] + h;
                float* p = &nmax[(long long)sDst[m] * WID + n];
                __hip_atomic_fetch_max(p, val, __ATOMIC_RELAXED,
                                       __HIP_MEMORY_SCOPE_AGENT);
            }
        }
        __syncthreads();
    }
}

// ---------------------------------------------------------------------------
// Node kernel: out = x + relu([x, fixfin(nmax)] @ W_mlp + b)   (K = 256)
// ---------------------------------------------------------------------------
#define KS_N 264   // padded k-stride for K=256
__global__ __launch_bounds__(256) void node_kernel(
    const float* __restrict__ x,  const float* __restrict__ nmax,
    const float* __restrict__ Wm, const float* __restrict__ bm,
    float* __restrict__ out)
{
    extern __shared__ char smem[];
    __bf16* sWt = (__bf16*)smem;                       // [128][KS_N] W^T bf16
    __bf16* sA  = (__bf16*)(smem + 128 * KS_N * 2);    // [16][KS_N]
    float*  sX  = (float*)(smem + 128 * KS_N * 2 + 16 * KS_N * 2); // [16][132]
    float*  sB  = sX + 16 * 132;                       // [128]

    const int tid = threadIdx.x;
    for (int i = tid; i < 256 * 128; i += 256) {
        int k = i >> 7, n = i & 127;
        sWt[n * KS_N + k] = (__bf16)Wm[k * 128 + n];
    }
    for (int i = tid; i < 128; i += 256) sB[i] = bm[i];
    __syncthreads();

    const int wave = tid >> 5, lane = tid & 31;
    const int nloc = lane & 15, half = lane >> 4;
    const int n    = (wave << 4) + nloc;

    const int nTiles = (NNODES + 15) / 16;
    for (int t = blockIdx.x; t < nTiles; t += gridDim.x) {
        const long long base = (long long)t * 16;

        for (int i = tid; i < 16 * 256; i += 256) {
            int m = i >> 8, c = i & 255;
            long long node = base + m;
            float v = 0.f;
            if (node < NNODES) {
                if (c < 128) {
                    v = x[node * WID + c];
                    sX[m * 132 + c] = v;
                } else {
                    float mx = nmax[node * WID + (c - 128)];
                    v = isfinite(mx) ? mx : 0.f;   // empty segments -> 0
                }
            } else if (c < 128) {
                sX[m * 132 + c] = 0.f;
            }
            sA[m * KS_N + c] = (__bf16)v;
        }
        __syncthreads();

        v8f c;
        #pragma unroll
        for (int r = 0; r < 8; ++r) c[r] = sB[n];

        #pragma unroll
        for (int kk = 0; kk < 8; ++kk) {
            const int kb = kk * 32;
            const __bf16* arow = &sA[(lane & 15) * KS_N + kb + (half << 3)];
            v8bf alo = *(const v8bf*)(arow);
            v8bf ahi = *(const v8bf*)(arow + 16);
            v16bf a = __builtin_shufflevector(alo, ahi,
                0,1,2,3,4,5,6,7,8,9,10,11,12,13,14,15);
            const __bf16* brow = &sWt[n * KS_N + kb + (half << 3)];
            v8bf blo = *(const v8bf*)(brow);
            v8bf bhi = *(const v8bf*)(brow + 16);
            v16bf b = __builtin_shufflevector(blo, bhi,
                0,1,2,3,4,5,6,7,8,9,10,11,12,13,14,15);
            c = __builtin_amdgcn_wmma_f32_16x16x32_bf16(
                    false, a, false, b, (short)0, c, false, false);
        }

        #pragma unroll
        for (int r = 0; r < 8; ++r) {
            int m = r + (half << 3);
            long long node = base + m;
            if (node < NNODES) {
                float h = c[r];
                h = h > 0.f ? h : 0.f;
                out[node * WID + n] = sX[m * 132 + n] + h;
            }
        }
        __syncthreads();
    }
}

// ---------------------------------------------------------------------------
extern "C" void kernel_launch(void* const* d_in, const int* in_sizes, int n_in,
                              void* d_out, int out_size, void* d_ws, size_t ws_size,
                              hipStream_t stream) {
    const float* x_p = (const float*)d_in[0];
    const int*   ei  = (const int*)  d_in[1];   // [2, E] row-major
    const float* efe = (const float*)d_in[2];   // [E, 2]
    const float* We  = (const float*)d_in[3];   // [130, 128]
    const float* be  = (const float*)d_in[4];   // [128]
    const float* Wm  = (const float*)d_in[5];   // [256, 128]
    const float* bm  = (const float*)d_in[6];   // [128]
    float*       out = (float*)d_out;
    float*       nmax = (float*)d_ws;           // [NNODES, 128] fp32 scratch

    const long long maxElems = (long long)NNODES * WID;
    init_max_kernel<<<2048, 256, 0, stream>>>(nmax, maxElems);

    const size_t edgeSmem = (size_t)128 * KS_E * 2 + (size_t)16 * KS_E * 2
                          + (size_t)16 * 132 * 4 + 32 * 4 + 128 * 4 + 16 * 4;
    edge_kernel<<<1024, 256, edgeSmem, stream>>>(x_p, ei, efe, We, be, nmax);

    const size_t nodeSmem = (size_t)128 * KS_N * 2 + (size_t)16 * KS_N * 2
                          + (size_t)16 * 132 * 4 + 128 * 4;
    node_kernel<<<512, 256, nodeSmem, stream>>>(x_p, nmax, Wm, bm, out);
}
